// EstimatorQNNHybrid_65481071407700
// MI455X (gfx1250) — compile-verified, matrix operand order checked by reading, not run on hardware
//
#include <hip/hip_runtime.h>
#include <hip/hip_bf16.h>
#include <math.h>

typedef __attribute__((ext_vector_type(16))) _Float16 v16h;
typedef __attribute__((ext_vector_type(8)))  float    v8f;

#define BATCH 8192

// ---- gfx1250 async direct-to-LDS copy (ASYNCcnt path), guarded ----------
// Probe result (round 3 stderr): __builtin_amdgcn_global_load_async_to_lds_b128
// exists, signature: (int __vector(4) __device__ *, int __vector(4) __shared__ *,
// imm offset, imm cpol).
#if defined(__gfx1250__) && __has_builtin(__builtin_amdgcn_global_load_async_to_lds_b128)
#define USE_ASYNC_LDS 1
typedef int v4i_ __attribute__((vector_size(16)));
typedef __attribute__((address_space(1))) v4i_ GlobalV4i;   // prints as __device__
typedef __attribute__((address_space(3))) v4i_ SharedV4i;   // prints as __shared__
#endif

// Copy 16 bytes (4 floats) global -> LDS. On gfx1250 this is one
// GLOBAL_LOAD_ASYNC_TO_LDS_B128 (no VGPR round-trip); flat->LDS address is
// the low 32 bits of the flat pointer.
__device__ __forceinline__ void g2l_b128(const float* g, float* l) {
#ifdef USE_ASYNC_LDS
  __builtin_amdgcn_global_load_async_to_lds_b128(
      (GlobalV4i*)(unsigned long long)g,
      (SharedV4i*)(unsigned int)(unsigned long long)l, 0, 0);
#else
  l[0] = g[0]; l[1] = g[1]; l[2] = g[2]; l[3] = g[3];
#endif
}

__device__ __forceinline__ void g2l_wait() {
#ifdef USE_ASYNC_LDS
#if __has_builtin(__builtin_amdgcn_s_wait_asynccnt)
  __builtin_amdgcn_s_wait_asynccnt(0);
#else
  asm volatile("s_wait_asynccnt 0x0" ::: "memory");
#endif
#endif
}

// Half index j (0..15) of a v16h A/B fragment -> K position, per CDNA5 ISA
// 16-bit 16x32 layout. g = lane/16.
__device__ __forceinline__ int kidx16(int j, int g) {
  return (j < 8) ? (g * 8 + j) : (16 + g * 8 + (j - 8));
}

// Templated on lane group G so k (and the k->(ic,ky,kx) decode) is fully
// compile-time constant: each LDS load folds its offset into the ds_load
// immediate. h1b points at h1p[py*16 + px]; h1p is [8][16][16] zero-padded.
template <int G>
__device__ __forceinline__ v16h conv2_a_frag(const float* h1b, const int ksBase) {
  v16h a;
  #pragma unroll
  for (int j = 0; j < 16; ++j) {
    const int k = ksBase + ((j < 8) ? (G * 8 + j) : (16 + G * 8 + (j - 8)));
    float v = 0.0f;
    if (k < 72) {
      const int ic = k / 9, r = k % 9, ky = r / 3, kx = r % 3;
      v = h1b[ic * 256 + ky * 16 + kx];
    }
    a[j] = (_Float16)v;
  }
  return a;
}

template <int G>
__device__ __forceinline__ v16h conv2_b_frag(const float* wn, const int ksBase) {
  v16h b;
  #pragma unroll
  for (int j = 0; j < 16; ++j) {
    const int k = ksBase + ((j < 8) ? (G * 8 + j) : (16 + G * 8 + (j - 8)));
    b[j] = (k < 72) ? (_Float16)wn[k] : (_Float16)0.0f;   // B[k][n] = w2[n][k]
  }
  return b;
}

// -------------------------------------------------------------------------
// K1: fused conv1(3x3,p1)+relu+maxpool2 (VALU) ->
//     conv2 as implicit GEMM on the matrix pipe:
//       M=196 positions (13 tiles of 16), N=16 out-ch, K=72 pad 96
//       -> 3 chained v_wmma_f32_16x16x32_f16 per M-tile
//     -> relu+maxpool2 -> avg[:6,:6]  =>  pooled[B,16]. One block per image.
// -------------------------------------------------------------------------
__global__ __launch_bounds__(256)
void conv_stack_kernel(const float* __restrict__ x,
                       const float* __restrict__ c1w, const float* __restrict__ c1b,
                       const float* __restrict__ c2w, const float* __restrict__ c2b,
                       float* __restrict__ pooled) {
  __shared__ __align__(16) float sx[784];       // 28x28 input
  __shared__ __align__(16) float sw1[72];       // 8x1x3x3
  __shared__ float sb1[8];
  __shared__ __align__(16) float sw2[1152];     // 16x(8*9)  k = ic*9+ky*3+kx
  __shared__ float sb2[16];
  __shared__ __align__(16) float h1p[2048];     // 8 x 16 x 16 zero-padded pool1
  __shared__ __align__(16) float h2full[3136];  // 16 x 196 relu(conv2)
  __shared__ float h2[784];                     // 16 x 7 x 7 post-pool

  const int b = blockIdx.x;
  const int tid = threadIdx.x;
  const float* xb = x + b * 784;

  // async stage: image + conv weights directly into LDS
  if (tid < 196) g2l_b128(xb + tid * 4, &sx[tid * 4]);       // 784 floats
  if (tid < 18)  g2l_b128(c1w + tid * 4, &sw1[tid * 4]);     // 72 floats
  for (int i = tid; i < 288; i += 256) g2l_b128(c2w + i * 4, &sw2[i * 4]); // 1152
  if (tid < 8)  sb1[tid] = c1b[tid];
  if (tid < 16) sb2[tid] = c2b[tid];
  for (int i = tid; i < 2048; i += 256) h1p[i] = 0.0f;       // zero border
  g2l_wait();
  __syncthreads();

  // conv1 + relu + 2x2 maxpool -> h1p interior [c][1+py][1+px]
  for (int p = tid; p < 1568; p += 256) {
    const int c = p / 196, pp = p % 196, py = pp / 14, px = pp % 14;
    const float* w = &sw1[c * 9];
    float m = 0.0f;
    for (int dy = 0; dy < 2; ++dy)
      for (int dx = 0; dx < 2; ++dx) {
        const int oy = 2 * py + dy, ox = 2 * px + dx;
        float acc = sb1[c];
        #pragma unroll
        for (int ky = 0; ky < 3; ++ky) {
          const int iy = oy + ky - 1;
          if (iy < 0 || iy >= 28) continue;
          #pragma unroll
          for (int kx = 0; kx < 3; ++kx) {
            const int ix = ox + kx - 1;
            if (ix < 0 || ix >= 28) continue;
            acc = fmaf(sx[iy * 28 + ix], w[ky * 3 + kx], acc);
          }
        }
        m = fmaxf(m, fmaxf(acc, 0.0f));
      }
    h1p[c * 256 + (py + 1) * 16 + (px + 1)] = m;
  }
  __syncthreads();

  // ---- conv2 via WMMA ----
  const int wv   = tid >> 5;
  const int lane = tid & 31;
  const int g    = lane >> 4;
  const int lr   = lane & 15;
  const float bias = sb2[lr];

  // B fragments (weights), built once per wave. Divergent build, no WMMA here.
  v16h bb0, bb1, bb2;
  {
    const float* wn = &sw2[lr * 72];    // n = lr (output channel)
    if (g == 0) {
      bb0 = conv2_b_frag<0>(wn, 0); bb1 = conv2_b_frag<0>(wn, 32); bb2 = conv2_b_frag<0>(wn, 64);
    } else {
      bb0 = conv2_b_frag<1>(wn, 0); bb1 = conv2_b_frag<1>(wn, 32); bb2 = conv2_b_frag<1>(wn, 64);
    }
  }

  for (int t = wv; t < 13; t += 8) {            // uniform per wave
    const int m  = t * 16 + lr;                 // A row = position index
    const int mc = (m < 196) ? m : 195;         // clamp: keep LDS reads valid
    const float* h1b = &h1p[(mc / 14) * 16 + (mc % 14)];

    v16h a0, a1, a2;
    if (g == 0) {
      a0 = conv2_a_frag<0>(h1b, 0); a1 = conv2_a_frag<0>(h1b, 32); a2 = conv2_a_frag<0>(h1b, 64);
    } else {
      a0 = conv2_a_frag<1>(h1b, 0); a1 = conv2_a_frag<1>(h1b, 32); a2 = conv2_a_frag<1>(h1b, 64);
    }
    // Reconverged: EXEC all-ones for WMMA.
    v8f c = {};
    c = __builtin_amdgcn_wmma_f32_16x16x32_f16(false, a0, false, bb0, (short)0, c, false, false);
    c = __builtin_amdgcn_wmma_f32_16x16x32_f16(false, a1, false, bb1, (short)0, c, false, false);
    c = __builtin_amdgcn_wmma_f32_16x16x32_f16(false, a2, false, bb2, (short)0, c, false, false);

    if (t < 12) {                               // uniform: branch-free stores
      #pragma unroll
      for (int r = 0; r < 8; ++r)
        h2full[lr * 196 + t * 16 + r + 8 * g] = fmaxf(c[r] + bias, 0.0f);
    } else {                                    // tail tile: rows 192..195 only
      #pragma unroll
      for (int r = 0; r < 8; ++r) {
        const int pos = 192 + r + 8 * g;
        if (pos < 196)
          h2full[lr * 196 + pos] = fmaxf(c[r] + bias, 0.0f);
      }
    }
  }
  __syncthreads();

  // 2x2 maxpool (inputs already relu'd >= 0)
  for (int q = tid; q < 784; q += 256) {
    const int c = q / 49, pp = q % 49, py = pp / 7, px = pp % 7;
    const float* hc = &h2full[c * 196];
    float m = 0.0f;
    #pragma unroll
    for (int dy = 0; dy < 2; ++dy)
      #pragma unroll
      for (int dx = 0; dx < 2; ++dx)
        m = fmaxf(m, hc[(2 * py + dy) * 14 + (2 * px + dx)]);
    h2[q] = m;
  }
  __syncthreads();

  // avg_pool2d(., 6): mean of top-left 6x6 of each 7x7 map
  if (tid < 16) {
    float s = 0.0f;
    for (int y = 0; y < 6; ++y)
      for (int xx = 0; xx < 6; ++xx) s += h2[tid * 49 + y * 7 + xx];
    pooled[b * 16 + tid] = s * (1.0f / 36.0f);
  }
}

// -------------------------------------------------------------------------
// K2: fc1 (relu) + fc2 via v_wmma_f32_16x16x32_f16.
//     One wave = 16 batch rows. fc1: K=16 (padded to 32), N=64 -> 4 WMMAs.
//     fc2: K=64 -> 2 chained WMMAs, N=4 padded to 16.
// -------------------------------------------------------------------------
__global__ __launch_bounds__(256)
void fc_wmma_kernel(const float* __restrict__ pooled,
                    const float* __restrict__ fc1w, const float* __restrict__ fc1b,
                    const float* __restrict__ fc2w, const float* __restrict__ fc2b,
                    float* __restrict__ out_raw) {
  __shared__ _Float16 fls[8 * 16 * 64];          // per-wave relu(fc1) tile, f16
  __shared__ __align__(16) float sfc1[1024];     // fc1_w [64][16]
  __shared__ __align__(16) float sfc2[256];      // fc2_w [4][64]
  __shared__ float sfb1[64];
  __shared__ float sfb2[4];
  const int tid  = threadIdx.x;
  const int wv   = tid >> 5;
  const int lane = tid & 31;
  const int g    = lane >> 4;
  const int lr   = lane & 15;
  const int rowbase = blockIdx.x * 128 + wv * 16;

  // async stage fc weights into LDS; overlaps with A-fragment global loads
  g2l_b128(fc1w + tid * 4, &sfc1[tid * 4]);               // 1024 floats, 256 thr
  if (tid < 64) g2l_b128(fc2w + tid * 4, &sfc2[tid * 4]); // 256 floats
  if (tid < 64) sfb1[tid] = fc1b[tid];
  if (tid < 4)  sfb2[tid] = fc2b[tid];

  // A fragment: pooled[16 rows x K16], zero-padded to K=32 (from global)
  v16h a;
  #pragma unroll
  for (int j = 0; j < 16; ++j) {
    const int k = kidx16(j, g);
    a[j] = (k < 16) ? (_Float16)pooled[(rowbase + lr) * 16 + k] : (_Float16)0.0f;
  }
  g2l_wait();
  __syncthreads();

  _Float16* fw = &fls[wv * 1024];
  #pragma unroll
  for (int nt = 0; nt < 4; ++nt) {
    const int n = nt * 16 + lr;          // output column (0..63)
    v16h bfrag;
    #pragma unroll
    for (int j = 0; j < 16; ++j) {
      const int k = kidx16(j, g);
      bfrag[j] = (k < 16) ? (_Float16)sfc1[n * 16 + k] : (_Float16)0.0f; // B[k][n]=fc1_w[n][k]
    }
    v8f c = {};
    c = __builtin_amdgcn_wmma_f32_16x16x32_f16(false, a, false, bfrag,
                                               (short)0, c, false, false);
    #pragma unroll
    for (int r = 0; r < 8; ++r) {
      const int m = r + g * 8;           // D layout: VGPR r, lane group g
      fw[m * 64 + n] = (_Float16)fmaxf(c[r] + sfb1[n], 0.0f);
    }
  }
  __syncthreads();

  // fc2: out = relu(f) @ fc2_w.T, K=64, N padded 4->16
  v8f c2 = {};
  #pragma unroll
  for (int kk = 0; kk < 2; ++kk) {
    v16h a2, b2;
    #pragma unroll
    for (int j = 0; j < 16; ++j) {
      const int k = kk * 32 + kidx16(j, g);
      a2[j] = fw[lr * 64 + k];
      b2[j] = (lr < 4) ? (_Float16)sfc2[lr * 64 + k] : (_Float16)0.0f;
    }
    c2 = __builtin_amdgcn_wmma_f32_16x16x32_f16(false, a2, false, b2,
                                                (short)0, c2, false, false);
  }
  if (lr < 4) {
    #pragma unroll
    for (int r = 0; r < 8; ++r) {
      const int m = r + g * 8;
      out_raw[(rowbase + m) * 4 + lr] = c2[r] + sfb2[lr];
    }
  }
}

// -------------------------------------------------------------------------
// K3/K5: batch mean + biased var for a [B,4] tensor. Single block.
// -------------------------------------------------------------------------
__global__ __launch_bounds__(256)
void bn_stats_kernel(const float* __restrict__ v, float* __restrict__ stats, int n) {
  __shared__ float ssum[256 * 4];
  __shared__ float ssq[256 * 4];
  const int tid = threadIdx.x;
  float s[4] = {0, 0, 0, 0}, q[4] = {0, 0, 0, 0};
  for (int i = tid; i < n; i += 256) {
    #pragma unroll
    for (int c = 0; c < 4; ++c) {
      const float x = v[i * 4 + c];
      s[c] += x; q[c] += x * x;
    }
  }
  #pragma unroll
  for (int c = 0; c < 4; ++c) { ssum[tid * 4 + c] = s[c]; ssq[tid * 4 + c] = q[c]; }
  __syncthreads();
  for (int off = 128; off > 0; off >>= 1) {
    if (tid < off) {
      #pragma unroll
      for (int c = 0; c < 4; ++c) {
        ssum[tid * 4 + c] += ssum[(tid + off) * 4 + c];
        ssq[tid * 4 + c]  += ssq[(tid + off) * 4 + c];
      }
    }
    __syncthreads();
  }
  if (tid == 0) {
    const float inv = 1.0f / (float)n;
    #pragma unroll
    for (int c = 0; c < 4; ++c) {
      const float m = ssum[c] * inv;
      stats[c] = m;
      stats[4 + c] = ssq[c] * inv - m * m;   // biased variance
    }
  }
}

// -------------------------------------------------------------------------
// K4: BN(out_raw) -> 4-qubit state-vector sim -> q_raw. One thread/sample;
//     16 complex amplitudes live in VGPRs.
// -------------------------------------------------------------------------
__global__ __launch_bounds__(256)
void quantum_kernel(const float* __restrict__ out_raw, const float* __restrict__ stats,
                    const float* __restrict__ ng, const float* __restrict__ nb,
                    const float* __restrict__ theta,
                    const float* __restrict__ ure, const float* __restrict__ uim,
                    float* __restrict__ out_bn, float* __restrict__ q_raw) {
  const int b = blockIdx.x * 256 + threadIdx.x;
  if (b >= BATCH) return;

  float o[4];
  #pragma unroll
  for (int c = 0; c < 4; ++c) {
    o[c] = ng[c] * (out_raw[b * 4 + c] - stats[c]) * rsqrtf(stats[4 + c] + 1e-5f) + nb[c];
    out_bn[b * 4 + c] = o[c];
  }

  float sr[16], si[16];
  #pragma unroll
  for (int i = 0; i < 16; ++i) { sr[i] = 0.0f; si[i] = 0.0f; }
  sr[0] = 1.0f;

  auto ap1 = [&](int w, float u00r, float u00i, float u01r, float u01i,
                 float u10r, float u10i, float u11r, float u11i) {
    const int m = 1 << (3 - w);
    #pragma unroll
    for (int i = 0; i < 16; ++i) {
      if (i & m) continue;
      const int j = i | m;
      const float ar = sr[i], ai = si[i], br = sr[j], bi = si[j];
      sr[i] = u00r * ar - u00i * ai + u01r * br - u01i * bi;
      si[i] = u00r * ai + u00i * ar + u01r * bi + u01i * br;
      sr[j] = u10r * ar - u10i * ai + u11r * br - u11i * bi;
      si[j] = u10r * ai + u10i * ar + u11r * bi + u11i * br;
    }
  };
  auto apc = [&](int cw, int tw, float u00r, float u00i, float u01r, float u01i,
                 float u10r, float u10i, float u11r, float u11i) {
    const int cm = 1 << (3 - cw), tm = 1 << (3 - tw);
    #pragma unroll
    for (int i = 0; i < 16; ++i) {
      if (!(i & cm) || (i & tm)) continue;
      const int j = i | tm;
      const float ar = sr[i], ai = si[i], br = sr[j], bi = si[j];
      sr[i] = u00r * ar - u00i * ai + u01r * br - u01i * bi;
      si[i] = u00r * ai + u00i * ar + u01r * bi + u01i * br;
      sr[j] = u10r * ar - u10i * ai + u11r * br - u11i * bi;
      si[j] = u10r * ai + u10i * ar + u11r * bi + u11i * br;
    }
  };

  // encoder '4x4_ryzxy': rounds RY,RZ,RX,RY over wires 0..3, angles cycled
  #pragma unroll
  for (int k = 0; k < 16; ++k) {
    const int w = k & 3;
    float sh, ch;
    sincosf(o[w] * 0.5f, &sh, &ch);
    const int r = k >> 2;
    if (r == 1)      ap1(w, ch, -sh, 0.f, 0.f, 0.f, 0.f, ch, sh);      // RZ
    else if (r == 2) ap1(w, ch, 0.f, 0.f, -sh, 0.f, -sh, ch, 0.f);     // RX
    else             ap1(w, ch, 0.f, -sh, 0.f, sh, 0.f, ch, 0.f);      // RY
  }

  // fixed 16x16 unitary: new_j = sum_i U[j][i]*s_i
  {
    float tr[16], ti[16];
    for (int j = 0; j < 16; ++j) {
      float ar = 0.f, ai = 0.f;
      #pragma unroll
      for (int i = 0; i < 16; ++i) {
        const float urv = ure[j * 16 + i], uiv = uim[j * 16 + i];
        ar += urv * sr[i] - uiv * si[i];
        ai += urv * si[i] + uiv * sr[i];
      }
      tr[j] = ar; ti[j] = ai;
    }
    #pragma unroll
    for (int i = 0; i < 16; ++i) { sr[i] = tr[i]; si[i] = ti[i]; }
  }

  float sh, ch;
  sincosf(theta[0] * 0.5f, &sh, &ch); ap1(0, ch, 0.f, 0.f, -sh, 0.f, -sh, ch, 0.f); // RX w0
  sincosf(theta[1] * 0.5f, &sh, &ch); ap1(1, ch, 0.f, -sh, 0.f, sh, 0.f, ch, 0.f);  // RY w1
  sincosf(theta[2] * 0.5f, &sh, &ch); ap1(2, ch, -sh, 0.f, 0.f, 0.f, 0.f, ch, sh);  // RZ w2
  sincosf(theta[3] * 0.5f, &sh, &ch); apc(0, 3, ch, 0.f, 0.f, -sh, 0.f, -sh, ch, 0.f); // CRX 0->3
  const float is2 = 0.70710678118654752440f;
  ap1(3, is2, 0.f, is2, 0.f, is2, 0.f, -is2, 0.f);                        // H w3
  ap1(2, 0.5f, 0.5f, 0.5f, -0.5f, 0.5f, -0.5f, 0.5f, 0.5f);               // SX w2
  apc(3, 0, 0.f, 0.f, 1.f, 0.f, 1.f, 0.f, 0.f, 0.f);                      // CX 3->0

  float q[4] = {0, 0, 0, 0};
  #pragma unroll
  for (int i = 0; i < 16; ++i) {
    const float p = sr[i] * sr[i] + si[i] * si[i];
    #pragma unroll
    for (int w = 0; w < 4; ++w)
      q[w] += ((i >> (3 - w)) & 1) ? -p : p;
  }
  #pragma unroll
  for (int c = 0; c < 4; ++c) q_raw[b * 4 + c] = q[c];
}

// -------------------------------------------------------------------------
// K6: d_out = out_bn + BN(q_raw)
// -------------------------------------------------------------------------
__global__ __launch_bounds__(256)
void final_kernel(const float* __restrict__ out_bn, const float* __restrict__ q_raw,
                  const float* __restrict__ stats,
                  const float* __restrict__ qg, const float* __restrict__ qb,
                  float* __restrict__ dout) {
  const int i = blockIdx.x * 256 + threadIdx.x;
  if (i >= BATCH * 4) return;
  const int c = i & 3;
  dout[i] = out_bn[i] +
            qg[c] * (q_raw[i] - stats[c]) * rsqrtf(stats[4 + c] + 1e-5f) + qb[c];
}

extern "C" void kernel_launch(void* const* d_in, const int* in_sizes, int n_in,
                              void* d_out, int out_size, void* d_ws, size_t ws_size,
                              hipStream_t stream) {
  const float* x    = (const float*)d_in[0];
  const float* c1w  = (const float*)d_in[1];
  const float* c1b  = (const float*)d_in[2];
  const float* c2w  = (const float*)d_in[3];
  const float* c2b  = (const float*)d_in[4];
  const float* fc1w = (const float*)d_in[5];
  const float* fc1b = (const float*)d_in[6];
  const float* fc2w = (const float*)d_in[7];
  const float* fc2b = (const float*)d_in[8];
  const float* ng   = (const float*)d_in[9];
  const float* nb   = (const float*)d_in[10];
  const float* th   = (const float*)d_in[11];
  const float* ure  = (const float*)d_in[12];
  const float* uim  = (const float*)d_in[13];
  const float* qg   = (const float*)d_in[14];
  const float* qb   = (const float*)d_in[15];

  float* ws      = (float*)d_ws;
  float* pooled  = ws;                    // 8192*16
  float* out_raw = pooled + BATCH * 16;   // 8192*4
  float* out_bn  = out_raw + BATCH * 4;   // 8192*4
  float* q_raw   = out_bn + BATCH * 4;    // 8192*4
  float* stats1  = q_raw + BATCH * 4;     // 8
  float* stats2  = stats1 + 8;            // 8

  conv_stack_kernel<<<BATCH, 256, 0, stream>>>(x, c1w, c1b, c2w, c2b, pooled);
  fc_wmma_kernel<<<BATCH / 128, 256, 0, stream>>>(pooled, fc1w, fc1b, fc2w, fc2b, out_raw);
  bn_stats_kernel<<<1, 256, 0, stream>>>(out_raw, stats1, BATCH);
  quantum_kernel<<<BATCH / 256, 256, 0, stream>>>(out_raw, stats1, ng, nb, th, ure, uim,
                                                  out_bn, q_raw);
  bn_stats_kernel<<<1, 256, 0, stream>>>(q_raw, stats2, BATCH);
  final_kernel<<<(BATCH * 4 + 255) / 256, 256, 0, stream>>>(out_bn, q_raw, stats2, qg, qb,
                                                            (float*)d_out);
}
